// VectorizedNPU_44135083933808
// MI455X (gfx1250) — matrix operand order repaired and checked by reading, not compiled, not run on hardware
//
#include <hip/hip_runtime.h>

// ---------------- problem constants ----------------
#define B_    16
#define NPUS  1024
#define MEMS  4096
#define MD    128
#define RS    256
#define INS   512
#define GRUIN 640
#define MMUO  644
#define MMUOP 656   // 644 padded to 41 tiles of 16

// output slice offsets (floats) in d_out, reference return order
static constexpr size_t O_OUT     = 0;                     // (16,1024,512)
static constexpr size_t O_NEWREG  = 8388608;               // (16,1024,256)
static constexpr size_t O_WKEY    = 12582912;              // (16,1024,128)
static constexpr size_t O_WVAL    = 14680064;              // (16,1024,128)
static constexpr size_t O_WGATE   = 16777216;              // (16,1024,1)
static constexpr size_t O_SSDKEY  = 16793600;              // (16,1024,128)
static constexpr size_t O_SSDVAL  = 18890752;              // (16,1024,128)
static constexpr size_t O_SSDGATE = 20987904;              // (16,1024,3)

// workspace offsets (bytes)
static constexpr size_t OFF_QB     = 0;          // q bf16        4 MiB
static constexpr size_t OFF_READB  = 4194304;    // read bf16     4 MiB
static constexpr size_t OFF_NREGB  = 8388608;    // new_regs bf16 8 MiB
static constexpr size_t OFF_REGB   = 16777216;   // regs bf16     8 MiB
static constexpr size_t OFF_MEMB   = 25165824;   // mem bf16     16 MiB
static constexpr size_t OFF_MEMT   = 41943040;   // memT bf16    16 MiB
static constexpr size_t OFF_INB    = 58720256;   // input bf16   16 KiB
static constexpr size_t OFF_WMMUT  = 58736640;   // [256][656]
static constexpr size_t OFF_WIHT   = 59072512;   // [640][768]
static constexpr size_t OFF_WHHT   = 60055552;   // [256][768]
static constexpr size_t OFF_WOUTT  = 60448768;   // [256][512]

typedef __attribute__((ext_vector_type(16))) __bf16 v16bf;
typedef __attribute__((ext_vector_type(8)))  float  v8f;

__device__ __forceinline__ unsigned short f2bf(float x) {
  union { float f; unsigned u; } v; v.f = x;
  unsigned u = v.u;
  u += 0x7FFFu + ((u >> 16) & 1u);      // round-to-nearest-even
  return (unsigned short)(u >> 16);
}

// two 16B chunks -> one A/B fragment (32 bytes = 16 bf16)
__device__ __forceinline__ v16bf load_pair(const unsigned short* p0,
                                           const unsigned short* p1) {
  union { uint4 q[2]; v16bf v; } f;
  f.q[0] = *(const uint4*)p0;
  f.q[1] = *(const uint4*)p1;
  return f.v;
}
__device__ __forceinline__ v16bf load32(const unsigned short* p) {
  union { uint4 q[2]; v16bf v; } f;
  f.q[0] = *(const uint4*)p;
  f.q[1] = *(const uint4*)(p + 8);
  return f.v;
}

__device__ __forceinline__ v8f wmma_bf16(v16bf a, v16bf b, v8f c) {
  // (neg_a, A, neg_b, B, c_mod, C, reuse_a, reuse_b)
  return __builtin_amdgcn_wmma_f32_16x16x32_bf16(false, a, false, b,
                                                 (short)0, c, false, false);
}

// A-fragment loader from a row-major bf16 matrix (16 rows starting at row0,
// K-chunk kc of 32): lane L -> row L&15, K = kbase + {0..7, 16..23}
__device__ __forceinline__ v16bf loadA(const unsigned short* base, int ld,
                                       int row0, int kc, int lane) {
  int kbase = (lane >> 4) * 8;
  const unsigned short* p =
      base + (size_t)(row0 + (lane & 15)) * ld + kc * 32 + kbase;
  return load_pair(p, p + 16);
}
__device__ __forceinline__ v16bf loadBs(const unsigned short* base, int ld,
                                        int col, int kc, int lane) {
  union { unsigned short s[16]; v16bf v; } f;
  int kbase = kc * 32 + (lane >> 4) * 8;
  for (int s = 0; s < 16; ++s) {
    int k = kbase + (s < 8 ? s : 8 + s);
    f.s[s] = base[(size_t)k * ld + col];
  }
  return f.v;
}

// ---------------- prep kernels ----------------
__global__ void k_cast(const float* __restrict__ in, unsigned short* out, int n) {
  int i = blockIdx.x * blockDim.x + threadIdx.x;
  if (i < n) out[i] = f2bf(in[i]);
}

// in: [z][R][C] f32  ->  out: [z][C][Rpad] bf16  (zero pad rows >= R)
__global__ void k_transpose(const float* __restrict__ in, unsigned short* out,
                            int R, int C, int Rpad) {
  size_t z = blockIdx.z;
  int i = blockIdx.x * blockDim.x + threadIdx.x;
  if (i >= C * Rpad) return;
  int c = i / Rpad, r = i % Rpad;
  float v = (r < R) ? in[z * (size_t)R * C + (size_t)r * C + c] : 0.f;
  out[z * (size_t)C * Rpad + i] = f2bf(v);
}

// ---------------- stage 1: mmu = regs @ WmmuT, slice + route ----------------
__global__ void k_mmu(const unsigned short* __restrict__ regb,
                      const unsigned short* __restrict__ WmmuT,
                      const float* __restrict__ b_mmu,
                      unsigned short* qb, float* out) {
  int lane = threadIdx.x;
  int c0 = blockIdx.x * 16, row0 = blockIdx.y * 16;
  int hi = lane >> 4, ln = lane & 15;
  v8f acc = {};
  for (int kc = 0; kc < 8; ++kc) {
    v16bf a = loadA(regb, RS, row0, kc, lane);
    v16bf b = loadBs(WmmuT, MMUOP, c0 + ln, kc, lane);
    acc = wmma_bf16(a, b, acc);
  }
  int col = c0 + ln;
  float bias = (col < MMUO) ? b_mmu[col] : 0.f;
  for (int r = 0; r < 8; ++r) {
    int row = row0 + r + hi * 8;           // == b*NPUS + n
    float v = acc[r] + bias;
    if (col < MD)            qb[(size_t)row * MD + col] = f2bf(v);
    else if (col < 2 * MD)   out[O_WKEY   + (size_t)row * MD + (col - MD)]      = v;
    else if (col < 3 * MD)   out[O_WVAL   + (size_t)row * MD + (col - 2 * MD)]  = v;
    else if (col == 3 * MD)  out[O_WGATE  + row] = 1.f / (1.f + __expf(-v));
    else if (col < 4 * MD + 1) out[O_SSDKEY + (size_t)row * MD + (col - 385)]   = v;
    else if (col < 5 * MD + 1) out[O_SSDVAL + (size_t)row * MD + (col - 513)]   = v;
    else if (col < MMUO)     out[O_SSDGATE + (size_t)row * 3 + (col - 641)]     = v;
  }
}

__global__ void k_gate_softmax(float* g) {
  int i = blockIdx.x * blockDim.x + threadIdx.x;
  if (i >= B_ * NPUS) return;
  float a = g[i * 3], b = g[i * 3 + 1], c = g[i * 3 + 2];
  float m = fmaxf(a, fmaxf(b, c));
  float ea = __expf(a - m), eb = __expf(b - m), ec = __expf(c - m);
  float s = 1.f / (ea + eb + ec);
  g[i * 3] = ea * s; g[i * 3 + 1] = eb * s; g[i * 3 + 2] = ec * s;
}

// ------------- stage 2: flash attention, one wave = 16 query rows -------------
__global__ void k_attn(const unsigned short* __restrict__ qb,
                       const unsigned short* __restrict__ memb,
                       const unsigned short* __restrict__ memT,
                       unsigned short* readb) {
  __shared__ unsigned short Pl[16 * 32];   // P tile bounce buffer (C -> A layout)
  int lane = threadIdx.x;
  int bb = blockIdx.y;
  int rowg0 = bb * NPUS + blockIdx.x * 16;
  int hi = lane >> 4, ln = lane & 15, kbase = hi * 8;

  v16bf qa[4];
  for (int kc = 0; kc < 4; ++kc) qa[kc] = loadA(qb, MD, rowg0, kc, lane);

  float m_i[8], l_i[8], alpha[8];
  v8f acc[8];
  for (int r = 0; r < 8; ++r) { m_i[r] = -1e30f; l_i[r] = 0.f; }
  for (int nt = 0; nt < 8; ++nt) acc[nt] = (v8f){};

  const float scale = 0.0883883476483184f;   // 1/sqrt(128)

  for (int it = 0; it < MEMS / 32; ++it) {
    int mbase = it * 32;
    v8f s0 = {}, s1 = {};
    for (int kc = 0; kc < 4; ++kc) {
      s0 = wmma_bf16(qa[kc], loadA(memb + (size_t)bb * MEMS * MD, MD, mbase, kc, lane), s0);
      s1 = wmma_bf16(qa[kc], loadA(memb + (size_t)bb * MEMS * MD, MD, mbase + 16, kc, lane), s1);
    }
    for (int r = 0; r < 8; ++r) {
      float x0 = s0[r] * scale, x1 = s1[r] * scale;
      float mx = fmaxf(x0, x1);
      for (int off = 8; off; off >>= 1) mx = fmaxf(mx, __shfl_xor(mx, off, 16));
      float nm = fmaxf(m_i[r], mx);
      float al = __expf(m_i[r] - nm);
      float p0 = __expf(x0 - nm), p1 = __expf(x1 - nm);
      float sm = p0 + p1;
      for (int off = 8; off; off >>= 1) sm += __shfl_xor(sm, off, 16);
      l_i[r] = l_i[r] * al + sm;
      m_i[r] = nm;
      alpha[r] = al;
      int M = r + hi * 8;
      Pl[M * 32 + ln]      = f2bf(p0);
      Pl[M * 32 + 16 + ln] = f2bf(p1);
    }
    for (int nt = 0; nt < 8; ++nt)
      for (int r = 0; r < 8; ++r) acc[nt][r] *= alpha[r];
    const unsigned short* pp = &Pl[ln * 32 + kbase];
    v16bf pa = load_pair(pp, pp + 16);           // P as 16x32 A fragment
    for (int nt = 0; nt < 8; ++nt) {
      acc[nt] = wmma_bf16(pa, loadA(memT + (size_t)bb * MD * MEMS, MEMS, nt * 16, it, lane), acc[nt]);
    }
  }
  for (int r = 0; r < 8; ++r) l_i[r] = 1.f / l_i[r];
  for (int nt = 0; nt < 8; ++nt)
    for (int r = 0; r < 8; ++r) {
      int row = rowg0 + r + hi * 8;
      readb[(size_t)row * MD + nt * 16 + ln] = f2bf(acc[nt][r] * l_i[r]);
    }
}

// ------------- stage 3: GRU (gi + gh + gates), wave = 16x16 tile of 256 -------
__global__ void k_gru(const unsigned short* __restrict__ inb,
                      const unsigned short* __restrict__ readb,
                      const unsigned short* __restrict__ regb,
                      const unsigned short* __restrict__ WihT,
                      const unsigned short* __restrict__ WhhT,
                      const float* __restrict__ b_ih, const float* __restrict__ b_hh,
                      const float* __restrict__ regs_f32,
                      float* out_newregs, unsigned short* newregb) {
  int lane = threadIdx.x;
  int j0 = blockIdx.x * 16, row0 = blockIdx.y * 16;
  int bb = row0 >> 10;
  int hi = lane >> 4, ln = lane & 15, kbase = hi * 8;
  v8f gir = {}, giz = {}, gin = {}, ghr = {}, ghz = {}, ghn = {};
  for (int kc = 0; kc < 20; ++kc) {               // x = [input(512) | read(128)]
    v16bf a;
    if (kc < 16) {
      const unsigned short* p = inb + bb * INS + kc * 32 + kbase; // row-invariant
      a = load_pair(p, p + 16);
    } else {
      a = loadA(readb, MD, row0, kc - 16, lane);
    }
    gir = wmma_bf16(a, loadBs(WihT, 768, j0 + ln, kc, lane),        gir);
    giz = wmma_bf16(a, loadBs(WihT, 768, j0 + ln + RS, kc, lane),   giz);
    gin = wmma_bf16(a, loadBs(WihT, 768, j0 + ln + 2*RS, kc, lane), gin);
  }
  for (int kc = 0; kc < 8; ++kc) {
    v16bf a = loadA(regb, RS, row0, kc, lane);
    ghr = wmma_bf16(a, loadBs(WhhT, 768, j0 + ln, kc, lane),        ghr);
    ghz = wmma_bf16(a, loadBs(WhhT, 768, j0 + ln + RS, kc, lane),   ghz);
    ghn = wmma_bf16(a, loadBs(WhhT, 768, j0 + ln + 2*RS, kc, lane), ghn);
  }
  int j = j0 + ln;
  float bi_r = b_ih[j], bi_z = b_ih[j + RS], bi_n = b_ih[j + 2 * RS];
  float bh_r = b_hh[j], bh_z = b_hh[j + RS], bh_n = b_hh[j + 2 * RS];
  for (int r = 0; r < 8; ++r) {
    int row = row0 + r + hi * 8;
    float rr = 1.f / (1.f + __expf(-(gir[r] + bi_r + ghr[r] + bh_r)));
    float zz = 1.f / (1.f + __expf(-(giz[r] + bi_z + ghz[r] + bh_z)));
    float nn = tanhf(gin[r] + bi_n + rr * (ghn[r] + bh_n));
    float cur = regs_f32[(size_t)row * RS + j];
    float nv = (1.f - zz) * nn + zz * cur;
    out_newregs[(size_t)row * RS + j] = nv;
    newregb[(size_t)row * RS + j] = f2bf(nv);
  }
}

// ------------- stage 4: output = new_regs @ WoutT + b_out ----------------
__global__ void k_out(const unsigned short* __restrict__ newregb,
                      const unsigned short* __restrict__ WoutT,
                      const float* __restrict__ b_out, float* out0) {
  int lane = threadIdx.x;
  int c0 = blockIdx.x * 16, row0 = blockIdx.y * 16;
  int hi = lane >> 4, ln = lane & 15;
  v8f acc = {};
  for (int kc = 0; kc < 8; ++kc) {
    v16bf a = loadA(newregb, RS, row0, kc, lane);
    v16bf b = loadBs(WoutT, INS, c0 + ln, kc, lane);
    acc = wmma_bf16(a, b, acc);
  }
  int col = c0 + ln;
  float bo = b_out[col];
  for (int r = 0; r < 8; ++r) {
    int row = row0 + r + hi * 8;
    out0[(size_t)row * INS + col] = acc[r] + bo;
  }
}

extern "C" void kernel_launch(void* const* d_in, const int* in_sizes, int n_in,
                              void* d_out, int out_size, void* d_ws, size_t ws_size,
                              hipStream_t stream) {
  (void)in_sizes; (void)n_in; (void)out_size; (void)ws_size;
  const float* input  = (const float*)d_in[0];
  const float* regs   = (const float*)d_in[1];
  const float* mem    = (const float*)d_in[2];
  const float* W_ih   = (const float*)d_in[3];
  const float* b_ih   = (const float*)d_in[4];
  const float* W_hh   = (const float*)d_in[5];
  const float* b_hh   = (const float*)d_in[6];
  const float* W_mmu  = (const float*)d_in[7];
  const float* b_mmu  = (const float*)d_in[8];
  const float* W_out  = (const float*)d_in[9];
  const float* b_out  = (const float*)d_in[10];
  float* out = (float*)d_out;
  char* ws = (char*)d_ws;
  unsigned short* qb     = (unsigned short*)(ws + OFF_QB);
  unsigned short* readb  = (unsigned short*)(ws + OFF_READB);
  unsigned short* nregb  = (unsigned short*)(ws + OFF_NREGB);
  unsigned short* regb   = (unsigned short*)(ws + OFF_REGB);
  unsigned short* memb   = (unsigned short*)(ws + OFF_MEMB);
  unsigned short* memT   = (unsigned short*)(ws + OFF_MEMT);
  unsigned short* inb    = (unsigned short*)(ws + OFF_INB);
  unsigned short* WmmuT  = (unsigned short*)(ws + OFF_WMMUT);
  unsigned short* WihT   = (unsigned short*)(ws + OFF_WIHT);
  unsigned short* WhhT   = (unsigned short*)(ws + OFF_WHHT);
  unsigned short* WoutT  = (unsigned short*)(ws + OFF_WOUTT);

  // prep: bf16 casts + transposes
  k_cast<<<(B_*NPUS*RS + 255) / 256, 256, 0, stream>>>(regs, regb, B_*NPUS*RS);
  k_cast<<<(B_*INS + 255) / 256, 256, 0, stream>>>(input, inb, B_*INS);
  k_cast<<<(B_*MEMS*MD + 255) / 256, 256, 0, stream>>>(mem, memb, B_*MEMS*MD);
  k_transpose<<<dim3((MD*MEMS + 255) / 256, 1, B_), 256, 0, stream>>>(mem, memT, MEMS, MD, MEMS);
  k_transpose<<<dim3((RS*MMUOP + 255) / 256, 1, 1), 256, 0, stream>>>(W_mmu, WmmuT, MMUO, RS, MMUOP);
  k_transpose<<<dim3((GRUIN*768 + 255) / 256, 1, 1), 256, 0, stream>>>(W_ih, WihT, 768, GRUIN, 768);
  k_transpose<<<dim3((RS*768 + 255) / 256, 1, 1), 256, 0, stream>>>(W_hh, WhhT, 768, RS, 768);
  k_transpose<<<dim3((RS*INS + 255) / 256, 1, 1), 256, 0, stream>>>(W_out, WoutT, INS, RS, INS);

  // pipeline
  k_mmu<<<dim3(MMUOP / 16, B_*NPUS / 16), 32, 0, stream>>>(regb, WmmuT, b_mmu, qb, out);
  k_gate_softmax<<<(B_*NPUS + 255) / 256, 256, 0, stream>>>(out + O_SSDGATE);
  k_attn<<<dim3(NPUS / 16, B_), 32, 0, stream>>>(qb, memb, memT, readb);
  k_gru<<<dim3(RS / 16, B_*NPUS / 16), 32, 0, stream>>>(inb, readb, regb, WihT, WhhT,
                                                        b_ih, b_hh, regs,
                                                        out + O_NEWREG, nregb);
  k_out<<<dim3(INS / 16, B_*NPUS / 16), 32, 0, stream>>>(nregb, WoutT, b_out, out + O_OUT);
}
